// SelfAttn_83734682403228
// MI455X (gfx1250) — compile-verified, matrix operand order checked by reading, not run on hardware
//
#include <hip/hip_runtime.h>
#include <hip/hip_bf16.h>

// ---------------------------------------------------------------------------
// Self-attention (SAGAN) for MI455X / gfx1250, wave32, bf16 WMMA pipeline.
// B=4, C=512, N=64*64=4096, C8=64.
// ---------------------------------------------------------------------------

typedef __attribute__((ext_vector_type(16))) __bf16 v16bf;
typedef __attribute__((ext_vector_type(8)))  __bf16 v8bf;
typedef __attribute__((ext_vector_type(8)))  float  v8f;

static constexpr int kB  = 4;
static constexpr int kC  = 512;
static constexpr int kN  = 4096;   // 64*64
static constexpr int kC8 = 64;

// Load a 16x32 bf16 A/B fragment whose lane-row is contiguous in memory.
// Per ISA layout: lane l holds row (l&15); element i maps to
// K = (i<8 ? i : i+8) + 8*(l>>4)  -> two contiguous 8-element (16B) runs.
__device__ __forceinline__ v16bf ld_frag_row(const __bf16* rowbase, int k0, int kh) {
  const v8bf lo = *(const v8bf*)(rowbase + k0 + kh);
  const v8bf hi = *(const v8bf*)(rowbase + k0 + 16 + kh);
  v16bf r;
#pragma unroll
  for (int i = 0; i < 8; ++i) { r[i] = lo[i]; r[8 + i] = hi[i]; }
  return r;
}

// f32 weight/att row -> bf16 fragment (two 32B runs, packed cvt)
__device__ __forceinline__ v16bf ld_frag_row_f32(const float* rowbase, int k0, int kh) {
  float4 a0 = *(const float4*)(rowbase + k0 + kh);
  float4 a1 = *(const float4*)(rowbase + k0 + kh + 4);
  float4 a2 = *(const float4*)(rowbase + k0 + 16 + kh);
  float4 a3 = *(const float4*)(rowbase + k0 + 16 + kh + 4);
  v16bf r;
  r[0] = (__bf16)a0.x; r[1] = (__bf16)a0.y; r[2]  = (__bf16)a0.z; r[3]  = (__bf16)a0.w;
  r[4] = (__bf16)a1.x; r[5] = (__bf16)a1.y; r[6]  = (__bf16)a1.z; r[7]  = (__bf16)a1.w;
  r[8] = (__bf16)a2.x; r[9] = (__bf16)a2.y; r[10] = (__bf16)a2.z; r[11] = (__bf16)a2.w;
  r[12]= (__bf16)a3.x; r[13]= (__bf16)a3.y; r[14] = (__bf16)a3.z; r[15] = (__bf16)a3.w;
  return r;
}

// ---------------------------------------------------------------------------
// Kernel 1: channel projection  P[b] = W[M x C] * feat[b][C x N] + bias
// Register-blocked: each wave owns a 64(M) x 16(N) output strip so the
// expensive strided feature (B) fragment is shared by 4 WMMAs.
//   transposeOut=1 : Out[b][n][M]  (bf16, position-major; q and k^T)
//   transposeOut=0 : Out[b][m][N]  (bf16, channel-major;  v)
// ---------------------------------------------------------------------------
__global__ __launch_bounds__(128)
void proj_kernel(const float* __restrict__ In,    // [B][C][N] f32
                 const float* __restrict__ Wt,    // [M][C]    f32
                 const float* __restrict__ bias,  // [M]       f32
                 __bf16* __restrict__ Out,
                 int M, int transposeOut) {
  const int lane = threadIdx.x & 31;
  const int wave = threadIdx.x >> 5;
  const int b    = blockIdx.z;
  const int m0   = blockIdx.y * 64;                       // 4 subtiles of 16
  const int n0   = (blockIdx.x * 4 + wave) * 16;
  const int l15  = lane & 15;
  const int kh   = (lane >> 4) * 8;

  const float* Arow = Wt + (size_t)(m0 + l15) * kC;          // weight row (contig K)
  const float* Bcol = In + (size_t)b * kC * kN + n0 + l15;   // feature column (stride N)

  v8f acc[4] = {};
  for (int c0 = 0; c0 < kC; c0 += 32) {
    // B fragment: strided gather, coalesced across lanes for each K; shared 4x
    v16bf bfg;
#pragma unroll
    for (int i = 0; i < 16; ++i) {
      const int k = (i < 8 ? i : i + 8) + kh;
      bfg[i] = (__bf16)Bcol[(size_t)(c0 + k) * kN];
    }
#pragma unroll
    for (int s = 0; s < 4; ++s) {
      const v16bf af = ld_frag_row_f32(Arow + (size_t)s * 16 * kC, c0, kh);
      acc[s] = __builtin_amdgcn_wmma_f32_16x16x32_bf16(false, af, false, bfg,
                                                       (short)0, acc[s], false, false);
    }
  }
#pragma unroll
  for (int s = 0; s < 4; ++s) {
    // bias along the M (row) dimension of D
#pragma unroll
    for (int j = 0; j < 8; ++j) acc[s][j] += bias[m0 + s * 16 + j + kh];

    if (transposeOut) {
      // Out[b][n][M]: per lane n fixed, 8 consecutive m -> one 16B store
      __bf16* p = Out + ((size_t)b * kN + (n0 + l15)) * M + m0 + s * 16 + kh;
      v8bf t;
#pragma unroll
      for (int j = 0; j < 8; ++j) t[j] = (__bf16)acc[s][j];
      *(v8bf*)p = t;
    } else {
      // Out[b][m][N]: scatter 8 bf16 (stride N)
      __bf16* base = Out + (size_t)b * M * kN + n0 + l15;
#pragma unroll
      for (int j = 0; j < 8; ++j)
        base[(size_t)(m0 + s * 16 + j + kh) * kN] = (__bf16)acc[s][j];
    }
  }
}

// ---------------------------------------------------------------------------
// Kernel 2: energy + softmax, fused in LDS.
// One workgroup (8 waves) owns 16 query rows x all 4096 keys:
// 16*4096 f32 = 256KB energy buffer in LDS (WGP has 320KB).
// E = Q[16 x 64] * K^T[64 x 4096] via 2 bf16 WMMAs per 16x16 tile,
// then per-row softmax (wave32 shuffle reductions), f32 attention -> d_out.
// ---------------------------------------------------------------------------
__global__ __launch_bounds__(256)
void attn_kernel(const __bf16* __restrict__ Q,   // [B][N][C8]
                 const __bf16* __restrict__ KT,  // [B][N][C8]
                 float* __restrict__ Att) {      // [B][N][N]
  extern __shared__ float Esm[];                 // [16][4096]
  const int lane = threadIdx.x & 31;
  const int wave = threadIdx.x >> 5;             // 0..7
  const int b    = blockIdx.y;
  const int m0   = blockIdx.x * 16;
  const int l15  = lane & 15;
  const int kh   = (lane >> 4) * 8;

  // A fragments for the 16 query rows (K = C8 = 64 -> two K=32 chunks)
  const __bf16* qrow = Q + ((size_t)b * kN + m0 + l15) * kC8;
  const v16bf a0 = ld_frag_row(qrow, 0,  kh);
  const v16bf a1 = ld_frag_row(qrow, 32, kh);

  // Each wave sweeps 32 of the 256 key tiles
  for (int t = 0; t < 32; ++t) {
    const int n0 = (t * 8 + wave) * 16;
    const __bf16* krow = KT + ((size_t)b * kN + n0 + l15) * kC8;
    const v16bf b0 = ld_frag_row(krow, 0,  kh);
    const v16bf b1 = ld_frag_row(krow, 32, kh);
    v8f acc = {};
    acc = __builtin_amdgcn_wmma_f32_16x16x32_bf16(false, a0, false, b0, (short)0, acc, false, false);
    acc = __builtin_amdgcn_wmma_f32_16x16x32_bf16(false, a1, false, b1, (short)0, acc, false, false);
#pragma unroll
    for (int j = 0; j < 8; ++j)
      Esm[(size_t)(j + kh) * kN + n0 + l15] = acc[j];
  }
  __syncthreads();

  // Softmax: 2 rows per wave, float4 sweeps, wave32 shuffle reductions
  for (int rr = 0; rr < 2; ++rr) {
    const int r = wave * 2 + rr;
    float4* row4 = (float4*)(Esm + (size_t)r * kN);
    float mx = -3.402823466e+38f;
    for (int it = 0; it < 32; ++it) {
      float4 v = row4[it * 32 + lane];
      mx = fmaxf(mx, fmaxf(fmaxf(v.x, v.y), fmaxf(v.z, v.w)));
    }
#pragma unroll
    for (int off = 16; off; off >>= 1) mx = fmaxf(mx, __shfl_xor(mx, off, 32));
    float s = 0.f;
    for (int it = 0; it < 32; ++it) {
      float4 v = row4[it * 32 + lane];
      v.x = __expf(v.x - mx); v.y = __expf(v.y - mx);
      v.z = __expf(v.z - mx); v.w = __expf(v.w - mx);
      row4[it * 32 + lane] = v;
      s += v.x + v.y + v.z + v.w;
    }
#pragma unroll
    for (int off = 16; off; off >>= 1) s += __shfl_xor(s, off, 32);
    const float inv = 1.0f / s;
    float4* dst4 = (float4*)(Att + ((size_t)b * kN + m0 + r) * kN);
    for (int it = 0; it < 32; ++it) {
      float4 v = row4[it * 32 + lane];
      v.x *= inv; v.y *= inv; v.z *= inv; v.w *= inv;
      dst4[it * 32 + lane] = v;
    }
  }
}

// ---------------------------------------------------------------------------
// Kernel 3: out[b][c][m] = gamma * sum_n v[b][c][n]*att[b][m][n] + x[b][c][m]
// Dominant GEMM (68.7 GF). Register-blocked: each wave owns a 64(C) x 16(M)
// output strip so one att (B) fragment feeds 4 WMMAs; att traffic drops to
// ~2.1 GB and stays L2-resident (67 MB/batch << 192 MB L2).
// ---------------------------------------------------------------------------
__global__ __launch_bounds__(128)
void out_kernel(const __bf16* __restrict__ V,    // [B][C][N] bf16
                const float* __restrict__ Att,   // [B][N][N] f32
                const float* __restrict__ X,     // [B][C][N] f32
                const float* __restrict__ gamma, // [1]
                float* __restrict__ Outp) {      // [B][C][N] f32
  const int lane = threadIdx.x & 31;
  const int wave = threadIdx.x >> 5;
  const int b    = blockIdx.z;
  const int c0   = blockIdx.y * 64;                       // 4 subtiles of 16
  const int m0   = (blockIdx.x * 4 + wave) * 16;
  const int l15  = lane & 15;
  const int kh   = (lane >> 4) * 8;

  const __bf16* vrow = V   + ((size_t)b * kC + c0 + l15) * kN;  // A lane row
  const float*  arow = Att + ((size_t)b * kN + m0 + l15) * kN;  // B lane "column" = att row

  v8f acc[4] = {};
  for (int n0 = 0; n0 < kN; n0 += 32) {
    // keep the two streaming operands ahead of the WMMA pipe
    __builtin_prefetch(arow + n0 + 1024, 0, 1);   // global_prefetch_b8
    __builtin_prefetch(vrow + n0 + 2048, 0, 1);
    // B fragment from f32 attention: shared by all 4 c-subtiles
    const v16bf bfg = ld_frag_row_f32(arow, n0, kh);
#pragma unroll
    for (int s = 0; s < 4; ++s) {
      const v16bf af = ld_frag_row(vrow + (size_t)s * 16 * kN, n0, kh);
      acc[s] = __builtin_amdgcn_wmma_f32_16x16x32_bf16(false, af, false, bfg,
                                                       (short)0, acc[s], false, false);
    }
  }
  const float g = gamma[0];
#pragma unroll
  for (int s = 0; s < 4; ++s) {
#pragma unroll
    for (int j = 0; j < 8; ++j) {
      const size_t idx = ((size_t)b * kC + c0 + s * 16 + j + kh) * kN + m0 + l15;
      Outp[idx] = g * acc[s][j] + X[idx];
    }
  }
}

// ---------------------------------------------------------------------------
extern "C" void kernel_launch(void* const* d_in, const int* in_sizes, int n_in,
                              void* d_out, int out_size, void* d_ws, size_t ws_size,
                              hipStream_t stream) {
  const float* x     = (const float*)d_in[0];
  const float* y     = (const float*)d_in[1];
  const float* wq    = (const float*)d_in[2];
  const float* bq    = (const float*)d_in[3];
  const float* wk    = (const float*)d_in[4];
  const float* bk    = (const float*)d_in[5];
  const float* wv    = (const float*)d_in[6];
  const float* bv    = (const float*)d_in[7];
  const float* gamma = (const float*)d_in[8];

  float* out = (float*)d_out;                       // [B][C][N]
  float* att = out + (size_t)kB * kC * kN;          // [B][N][N]

  // Workspace: bf16 operands, 20 MB total
  __bf16* qws = (__bf16*)d_ws;                      // [B][N][C8]  2MB
  __bf16* kws = qws + (size_t)kB * kN * kC8;        // [B][N][C8]  2MB
  __bf16* vws = kws + (size_t)kB * kN * kC8;        // [B][C][N]  16MB

  const dim3 blk128(128);
  // q and k^T projections (M = C8 = 64 -> one 64-row strip), position-major
  proj_kernel<<<dim3(kN / 64, kC8 / 64, kB), blk128, 0, stream>>>(y, wq, bq, qws, kC8, 1);
  proj_kernel<<<dim3(kN / 64, kC8 / 64, kB), blk128, 0, stream>>>(y, wk, bk, kws, kC8, 1);
  // v projection (M = C = 512 -> 8 strips), channel-major
  proj_kernel<<<dim3(kN / 64, kC / 64, kB), blk128, 0, stream>>>(x, wv, bv, vws, kC, 0);
  // fused energy + softmax: 256KB dynamic LDS per workgroup
  attn_kernel<<<dim3(kN / 16, kB), dim3(256), 16 * kN * sizeof(float), stream>>>(qws, kws, att);
  // output GEMM + gamma/residual epilogue, 64x16 per wave
  out_kernel<<<dim3(kN / 64, kC / 64, kB), blk128, 0, stream>>>(vws, att, x, gamma, out);
}
